// SimpleResNet10_86457691669047
// MI455X (gfx1250) — matrix operand order repaired and checked by reading, not compiled, not run on hardware
//
#include <hip/hip_runtime.h>
#include <cstdint>

// ---------------- CDNA5 WMMA operand types (wave32) ----------------
typedef __attribute__((ext_vector_type(16))) __bf16 v16bf;
typedef __attribute__((ext_vector_type(8)))  float  v8f;

#if defined(__HIP_DEVICE_COMPILE__) && __has_builtin(__builtin_amdgcn_global_load_async_to_lds_b128)
#define ASYNC_A 1
typedef int v4i_g __attribute__((vector_size(4 * sizeof(int))));
typedef __attribute__((address_space(1))) v4i_g* as1_v4i;   // global int4*
typedef __attribute__((address_space(3))) v4i_g* as3_v4i;   // LDS int4*
#endif

namespace {
constexpr int   BATCH  = 2;
constexpr int   NPTS   = 8192;
constexpr int   NCLS   = 5;
constexpr float BN_EPS = 1e-4f;
}

// ---------------- utility kernels ----------------
__global__ void k_zero(float* __restrict__ p, long n) {
  long i = (long)blockIdx.x * blockDim.x + threadIdx.x;
  if (i < n) p[i] = 0.0f;
}

__global__ void k_add(const float* __restrict__ a, const float* __restrict__ b,
                      float* __restrict__ o, long n) {
  long i = (long)blockIdx.x * blockDim.x + threadIdx.x;
  if (i < n) o[i] = a[i] + b[i];
}

// scatter points into dense grid + mask
__global__ void k_scatter(const float* __restrict__ blob, float* __restrict__ x,
                          float* __restrict__ mask) {
  int i = blockIdx.x * blockDim.x + threadIdx.x;
  if (i >= NPTS) return;
  const float* p = blob + (long)i * 5;
  int c0 = (int)p[0], c1 = (int)p[1], c2 = (int)p[2], b = (int)p[3];
  float f = p[4];
  long idx = (((long)b * 128 + c0) * 128 + c1) * 128 + c2;
  atomicAdd(&x[idx], f);
  mask[idx] = 1.0f;
}

// fused conv0 (3x3x3, 1->16, SAME, masked) + 2x2x2 masked maxpool to 64^3
__global__ void k_conv0_pool(const float* __restrict__ x, const float* __restrict__ mask,
                             const float* __restrict__ w0, float* __restrict__ y,
                             float* __restrict__ mout) {
  const int D = 64, Din = 128;
  int gv = blockIdx.x * blockDim.x + threadIdx.x;
  int nv = BATCH * D * D * D;
  if (gv >= nv) return;
  int b = gv / (D * D * D);
  int r = gv - b * D * D * D;
  int oz = r / (D * D); r -= oz * D * D;
  int oy = r / D;
  int ox = r - oy * D;

  float best[16];
#pragma unroll
  for (int c = 0; c < 16; ++c) best[c] = -1e30f;
  int any = 0;

  for (int dz = 0; dz < 2; ++dz)
    for (int dy = 0; dy < 2; ++dy)
      for (int dx = 0; dx < 2; ++dx) {
        int cz = 2 * oz + dz, cy = 2 * oy + dy, cx = 2 * ox + dx;
        long ci = (((long)b * Din + cz) * Din + cy) * Din + cx;
        if (mask[ci] != 0.0f) {
          any = 1;
          float acc[16];
#pragma unroll
          for (int c = 0; c < 16; ++c) acc[c] = 0.0f;
          for (int tz = 0; tz < 3; ++tz)
            for (int ty = 0; ty < 3; ++ty)
              for (int tx = 0; tx < 3; ++tx) {
                int iz = cz + tz - 1, iy = cy + ty - 1, ix = cx + tx - 1;
                if ((unsigned)iz < 128u && (unsigned)iy < 128u && (unsigned)ix < 128u) {
                  float v = x[(((long)b * Din + iz) * Din + iy) * Din + ix];
                  int tap = (tz * 3 + ty) * 3 + tx;
#pragma unroll
                  for (int c = 0; c < 16; ++c) acc[c] += v * w0[tap * 16 + c];
                }
              }
#pragma unroll
          for (int c = 0; c < 16; ++c) best[c] = fmaxf(best[c], acc[c]);
        }
      }
  mout[gv] = any ? 1.0f : 0.0f;
#pragma unroll
  for (int c = 0; c < 16; ++c) y[(long)gv * 16 + c] = any ? best[c] : 0.0f;
}

// mask pooling: max over win^3 window (VALID)
__global__ void k_pool_mask(const float* __restrict__ min_, float* __restrict__ mout,
                            int Din, int Dout, int win, int stride) {
  int gv = blockIdx.x * blockDim.x + threadIdx.x;
  int nv = BATCH * Dout * Dout * Dout;
  if (gv >= nv) return;
  int b = gv / (Dout * Dout * Dout);
  int r = gv - b * Dout * Dout * Dout;
  int oz = r / (Dout * Dout); r -= oz * Dout * Dout;
  int oy = r / Dout;
  int ox = r - oy * Dout;
  float m = 0.0f;
  for (int dz = 0; dz < win; ++dz)
    for (int dy = 0; dy < win; ++dy)
      for (int dx = 0; dx < win; ++dx) {
        int iz = oz * stride + dz, iy = oy * stride + dy, ix = ox * stride + dx;
        m = fmaxf(m, min_[(((long)b * Din + iz) * Din + iy) * Din + ix]);
      }
  mout[gv] = (m > 0.0f) ? 1.0f : 0.0f;
}

// masked BN statistics: stats[0]=cnt, stats[1..C]=sum(x*m), stats[1+C..2C]=sum(x*m*x)
__global__ void k_bn_stats(const float* __restrict__ x, const float* __restrict__ mask,
                           float* __restrict__ stats, long nvox, int C) {
  long total = (long)gridDim.x * blockDim.x;   // multiple of C (C divides 256)
  long tid   = (long)blockIdx.x * blockDim.x + threadIdx.x;
  int  c     = (int)(tid % C);                 // channel is stride-invariant
  float s = 0.0f, sq = 0.0f, cnt = 0.0f;
  long n = nvox * (long)C;
  for (long idx = tid; idx < n; idx += total) {
    long vox = idx / C;
    float m = mask[vox];
    float v = x[idx];
    float vm = v * m;
    s  += vm;
    sq += vm * v;
    if (c == 0) cnt += m;
  }
  atomicAdd(&stats[1 + c], s);
  atomicAdd(&stats[1 + C + c], sq);
  if (c == 0) atomicAdd(&stats[0], cnt);
}

__global__ void k_bn_relu(const float* __restrict__ x, const float* __restrict__ mask,
                          const float* __restrict__ stats, const float* __restrict__ g,
                          const float* __restrict__ bia, float* __restrict__ y,
                          long n, int C) {
  long i = (long)blockIdx.x * blockDim.x + threadIdx.x;
  if (i >= n) return;
  int c = (int)(i % C);
  long vox = i / C;
  float cnt  = fmaxf(stats[0], 1.0f);
  float mean = stats[1 + c] / cnt;
  float var  = stats[1 + C + c] / cnt - mean * mean;
  float v = (x[i] - mean) * rsqrtf(var + BN_EPS) * g[c] + bia[c];
  y[i] = fmaxf(v, 0.0f) * mask[vox];
}

// -------- implicit-GEMM masked conv using v_wmma_f32_16x16x32_bf16 --------
// M-tile: 16 output voxels (blockIdx.x); N: COUT/16 tiles, round-robin over the
// 4 waves; K = KS^3*CIN in chunks of 32.  A chunks staged via async-to-LDS
// (contiguous 128B per voxel when CIN>=32), per-(voxel,tap) base offsets
// precomputed once per tile.
template <int CIN, int COUT, int KS, int STRIDE, int PAD>
__global__ __launch_bounds__(128) void k_conv_wmma(
    const float* __restrict__ x, const float* __restrict__ w,
    const float* __restrict__ omask, float* __restrict__ out,
    int Din, int Dout) {
  constexpr int TAPS = KS * KS * KS;
  constexpr int K    = TAPS * CIN;
  constexpr int NT   = COUT / 16;          // N tiles
  const int nvox = BATCH * Dout * Dout * Dout;
  const int tile = blockIdx.x;
  const int tid  = threadIdx.x;
  const int lane = tid & 31;
  const int wave = tid >> 5;

  __shared__ float lds_af[16][36];         // A chunk, f32 (row stride 144B)
  __shared__ int   sbase[16][TAPS];        // per (voxel, tap) input base offset
  __shared__ float lds_mask[16];
  __shared__ int   lds_any;

  if (tid == 0) lds_any = 0;
  __syncthreads();
  if (tid < 16) {
    int gv = tile * 16 + tid;
    float m = (gv < nvox) ? omask[gv] : 0.0f;
    lds_mask[tid] = m;
    if (m != 0.0f) atomicOr(&lds_any, 1);
  }
  __syncthreads();
  if (lds_any == 0) {                      // fully-unmasked tile
    for (int e = tid; e < 16 * COUT; e += 128) {
      int m = e / COUT, nn = e - (e / COUT) * COUT;
      int gv = tile * 16 + m;
      if (gv < nvox) out[(long)gv * COUT + nn] = 0.0f;
    }
    return;
  }

  // precompute per-(voxel,tap) base offsets (element index into x, or -1)
  const int D2 = Dout * Dout;
  for (int e = tid; e < 16 * TAPS; e += 128) {
    int m = e / TAPS, tap = e - (e / TAPS) * TAPS;
    int gv = tile * 16 + m;
    int base = -1;
    if (gv < nvox) {
      int tx = tap % KS;
      int tt = tap / KS;
      int ty = tt % KS;
      int tz = tt / KS;
      int b  = gv / (Dout * D2);
      int r  = gv - b * Dout * D2;
      int oz = r / D2; r -= oz * D2;
      int oy = r / Dout;
      int ox = r - oy * Dout;
      int iz = oz * STRIDE + tz - PAD;
      int iy = oy * STRIDE + ty - PAD;
      int ix = ox * STRIDE + tx - PAD;
      if ((unsigned)iz < (unsigned)Din && (unsigned)iy < (unsigned)Din &&
          (unsigned)ix < (unsigned)Din)
        base = (((b * Din + iz) * Din + iy) * Din + ix) * CIN;
    }
    sbase[m][tap] = base;
  }

  v8f vz = {0.f, 0.f, 0.f, 0.f, 0.f, 0.f, 0.f, 0.f};
  v8f acc0 = vz, acc1 = vz, acc2 = vz, acc3 = vz;
  v8f* accs[4] = {&acc0, &acc1, &acc2, &acc3};
  const int half = (lane >> 4) & 1;

  for (int kc = 0; kc < K; kc += 32) {
    __syncthreads();
    if constexpr (CIN >= 32) {
      // one contiguous 128B run per voxel row: 1 x b128 per thread
      int m  = tid >> 3;                   // 8 threads per row
      int t0 = (tid & 7) * 4;
      int tap = kc / CIN;                  // constexpr shift
      int c0  = kc & (CIN - 1);
      int base = sbase[m][tap];
      if (base >= 0) {
#ifdef ASYNC_A
        __builtin_amdgcn_global_load_async_to_lds_b128(
            (as1_v4i)(uintptr_t)(x + base + c0 + t0),
            (as3_v4i)(uint32_t)(uintptr_t)&lds_af[m][t0], 0, 0);
#else
        const float4 v4 = *reinterpret_cast<const float4*>(x + base + c0 + t0);
        lds_af[m][t0 + 0] = v4.x;
        lds_af[m][t0 + 1] = v4.y;
        lds_af[m][t0 + 2] = v4.z;
        lds_af[m][t0 + 3] = v4.w;
#endif
      } else {
        lds_af[m][t0 + 0] = 0.0f;
        lds_af[m][t0 + 1] = 0.0f;
        lds_af[m][t0 + 2] = 0.0f;
        lds_af[m][t0 + 3] = 0.0f;
      }
#ifdef ASYNC_A
#if __has_builtin(__builtin_amdgcn_s_wait_asynccnt)
      __builtin_amdgcn_s_wait_asynccnt(0);
#else
      asm volatile("s_wait_asynccnt 0x0" ::: "memory");
#endif
#endif
    } else {
      // CIN == 16: chunk spans 2 taps
      for (int e = tid; e < 512; e += 128) {
        int m = e >> 5;
        int t = e & 31;
        int k = kc + t;
        float val = 0.0f;
        if (k < K) {
          int tap = k >> 4;
          int cin = k & 15;
          int base = sbase[m][tap];
          if (base >= 0) val = x[base + cin];
        }
        lds_af[m][t] = val;
      }
    }
    __syncthreads();

    // A fragment per ISA 16-bit 16x32 layout (convert f32 -> bf16 here)
    v16bf afrag;
    {
      int mrow = lane & 15;
#pragma unroll
      for (int j = 0; j < 16; ++j) {
        int v = j >> 1, p = j & 1;
        int koff = ((v < 4) ? 0 : 16) + half * 8 + (v & 3) * 2 + p;
        afrag[j] = (__bf16)lds_af[mrow][koff];
      }
    }

#pragma unroll
    for (int it = 0; it < 4; ++it) {
      const int nt = wave + it * 4;
      if (nt < NT) {
        v16bf bfrag;
        int ncol = nt * 16 + (lane & 15);
#pragma unroll
        for (int j = 0; j < 16; ++j) {
          int v = j >> 1, p = j & 1;
          int kk = kc + ((v < 4) ? 0 : 16) + half * 8 + (v & 3) * 2 + p;
          float bv = 0.0f;
          if (kk < K) bv = w[(long)kk * COUT + ncol];
          bfrag[j] = (__bf16)bv;
        }
        if (kc + 32 < K)                   // prefetch next weight chunk
          __builtin_prefetch(&w[(long)(kc + 32) * COUT + ncol], 0, 1);
        *accs[it] = __builtin_amdgcn_wmma_f32_16x16x32_bf16(
            false, afrag, false, bfrag, (short)0, *accs[it], false, false);
      }
    }
  }

  // write D with output-mask applied
#pragma unroll
  for (int it = 0; it < 4; ++it) {
    const int nt = wave + it * 4;
    if (nt < NT) {
      int ncol = nt * 16 + (lane & 15);
      v8f a = *accs[it];
#pragma unroll
      for (int r = 0; r < 8; ++r) {
        int m = r + half * 8;
        int gv = tile * 16 + m;
        if (gv < nvox)
          out[(long)gv * COUT + ncol] = a[r] * lds_mask[m];
      }
    }
  }
}

// spatial mean: x (B, nsp, C) -> pooled (B, C)
__global__ void k_mean(const float* __restrict__ x, float* __restrict__ pooled,
                       int nsp, int C) {
  int i = blockIdx.x * blockDim.x + threadIdx.x;
  if (i >= BATCH * C) return;
  int b = i / C, c = i - (i / C) * C;
  float s = 0.0f;
  for (int v = 0; v < nsp; ++v) s += x[((long)b * nsp + v) * C + c];
  pooled[i] = s / (float)nsp;
}

__global__ void k_linear(const float* __restrict__ pooled, const float* __restrict__ w,
                         const float* __restrict__ b, float* __restrict__ out, int C) {
  int i = blockIdx.x * blockDim.x + threadIdx.x;
  if (i >= BATCH * NCLS) return;
  int bb = i / NCLS, cls = i - (i / NCLS) * NCLS;
  float s = b[cls];
  for (int c = 0; c < C; ++c) s += pooled[bb * C + c] * w[c * NCLS + cls];
  out[i] = s;
}

// ---------------- host orchestration ----------------
extern "C" void kernel_launch(void* const* d_in, const int* in_sizes, int n_in,
                              void* d_out, int out_size, void* d_ws, size_t ws_size,
                              hipStream_t stream) {
  (void)in_sizes; (void)n_in; (void)out_size; (void)ws_size;
  int idx = 0;
  const float* blob    = (const float*)d_in[idx++];
  const float* w_conv0 = (const float*)d_in[idx++];

  struct Blk { const float *bn0_g, *bn0_b, *w1, *res_w, *bn1_g, *bn1_b, *w2; };
  Blk blk[8];
  const int stage_n[4]      = {16, 32, 64, 128};
  const int stage_stride[4] = {1, 2, 2, 2};
  int bi = 0;
  for (int s = 0; s < 4; ++s)
    for (int rep = 0; rep < 2; ++rep) {
      Blk& p  = blk[bi++];
      p.bn0_g = (const float*)d_in[idx++];
      p.bn0_b = (const float*)d_in[idx++];
      p.w1    = (const float*)d_in[idx++];
      p.res_w = nullptr;
      if (rep == 0 && stage_stride[s] > 1) p.res_w = (const float*)d_in[idx++];
      p.bn1_g = (const float*)d_in[idx++];
      p.bn1_b = (const float*)d_in[idx++];
      p.w2    = (const float*)d_in[idx++];
    }
  const float* bnf_g  = (const float*)d_in[idx++];
  const float* bnf_b  = (const float*)d_in[idx++];
  const float* w_last = (const float*)d_in[idx++];
  const float* bnl_g  = (const float*)d_in[idx++];
  const float* bnl_b  = (const float*)d_in[idx++];
  const float* lin_w  = (const float*)d_in[idx++];
  const float* lin_b  = (const float*)d_in[idx++];

  float* ws = (float*)d_ws;
  size_t off = 0;
  auto carve = [&](size_t n) -> float* { float* p = ws + off; off += n; return p; };
  const size_t NV128 = (size_t)BATCH * 128 * 128 * 128;
  const size_t NV64  = (size_t)BATCH * 64 * 64 * 64;
  float* x128    = carve(NV128);
  float* mask128 = carve(NV128);
  float* bufX    = carve(NV64 * 16);
  float* bufH    = carve(NV64 * 16);
  float* bufR    = carve(NV64 * 16);
  float* bufT    = carve(NV64 * 16);
  float* mA      = carve(NV64);
  float* mB      = carve(NV64);
  float* stats   = carve(513);
  float* pooled  = carve((size_t)BATCH * 256);

  auto zero = [&](float* p, long n) {
    k_zero<<<dim3((unsigned)((n + 255) / 256)), dim3(256), 0, stream>>>(p, n);
  };
  auto bn = [&](const float* x, const float* m, const float* g, const float* b,
                float* y, long nvox, int C) {
    zero(stats, 513);
    k_bn_stats<<<dim3(256), dim3(256), 0, stream>>>(x, m, stats, nvox, C);
    long n = nvox * (long)C;
    k_bn_relu<<<dim3((unsigned)((n + 255) / 256)), dim3(256), 0, stream>>>(
        x, m, stats, g, b, y, n, C);
  };
  auto conv = [&](const float* x, const float* w, const float* om, float* out,
                  int Din, int Dout, int Cin, int Cout, int ks) {
    int nv = BATCH * Dout * Dout * Dout;
    dim3 g((unsigned)(nv / 16)), b(128);
    if      (Cin ==  16 && Cout ==  16 && ks == 3) k_conv_wmma< 16,  16, 3, 1, 1><<<g, b, 0, stream>>>(x, w, om, out, Din, Dout);
    else if (Cin ==  16 && Cout ==  32 && ks == 2) k_conv_wmma< 16,  32, 2, 2, 0><<<g, b, 0, stream>>>(x, w, om, out, Din, Dout);
    else if (Cin ==  32 && Cout ==  32 && ks == 3) k_conv_wmma< 32,  32, 3, 1, 1><<<g, b, 0, stream>>>(x, w, om, out, Din, Dout);
    else if (Cin ==  32 && Cout ==  64 && ks == 2) k_conv_wmma< 32,  64, 2, 2, 0><<<g, b, 0, stream>>>(x, w, om, out, Din, Dout);
    else if (Cin ==  64 && Cout ==  64 && ks == 3) k_conv_wmma< 64,  64, 3, 1, 1><<<g, b, 0, stream>>>(x, w, om, out, Din, Dout);
    else if (Cin ==  64 && Cout == 128 && ks == 2) k_conv_wmma< 64, 128, 2, 2, 0><<<g, b, 0, stream>>>(x, w, om, out, Din, Dout);
    else if (Cin == 128 && Cout == 128 && ks == 3) k_conv_wmma<128, 128, 3, 1, 1><<<g, b, 0, stream>>>(x, w, om, out, Din, Dout);
    else if (Cin == 128 && Cout == 256 && ks == 3) k_conv_wmma<128, 256, 3, 1, 0><<<g, b, 0, stream>>>(x, w, om, out, Din, Dout);
  };
  auto addk = [&](const float* a, const float* b, float* o, long n) {
    k_add<<<dim3((unsigned)((n + 255) / 256)), dim3(256), 0, stream>>>(a, b, o, n);
  };

  // scatter points into dense grid
  zero(x128, (long)NV128);
  zero(mask128, (long)NV128);
  k_scatter<<<dim3((NPTS + 255) / 256), dim3(256), 0, stream>>>(blob, x128, mask128);

  // conv0 + maxpool (fused) -> 64^3 x 16ch
  k_conv0_pool<<<dim3((unsigned)(NV64 / 128)), dim3(128), 0, stream>>>(
      x128, mask128, w_conv0, bufX, mA);

  float* x = bufX;
  float* mask = mA;
  float* maskO = mB;
  int D = 64, C = 16;
  bi = 0;
  for (int s = 0; s < 4; ++s) {
    const int n = stage_n[s], stride = stage_stride[s];
    for (int rep = 0; rep < 2; ++rep) {
      const Blk& p = blk[bi++];
      if (rep == 0 && stride > 1) {
        int Dn = D / 2;
        long nvIn  = (long)BATCH * D * D * D;
        long nvOut = (long)BATCH * Dn * Dn * Dn;
        bn(x, mask, p.bn0_g, p.bn0_b, bufT, nvIn, C);
        k_pool_mask<<<dim3((unsigned)((nvOut + 255) / 256)), dim3(256), 0, stream>>>(
            mask, maskO, D, Dn, 2, 2);
        conv(bufT, p.w1,    maskO, bufH, D, Dn, C, n, 2);
        conv(bufT, p.res_w, maskO, bufR, D, Dn, C, n, 2);
        bn(bufH, maskO, p.bn1_g, p.bn1_b, bufT, nvOut, n);
        conv(bufT, p.w2, maskO, bufH, Dn, Dn, n, n, 3);
        addk(bufH, bufR, x, nvOut * (long)n);
        D = Dn; C = n;
        float* t = mask; mask = maskO; maskO = t;
      } else {
        long nv = (long)BATCH * D * D * D;
        bn(x, mask, p.bn0_g, p.bn0_b, bufT, nv, C);
        conv(bufT, p.w1, mask, bufH, D, D, C, n, 3);
        bn(bufH, mask, p.bn1_g, p.bn1_b, bufT, nv, n);
        conv(bufT, p.w2, mask, bufH, D, D, n, n, 3);
        addk(bufH, x, x, nv * (long)n);   // residual: res == x
        C = n;
      }
    }
  }

  // head: D==8, C==128
  long nv8 = (long)BATCH * 8 * 8 * 8;
  bn(x, mask, bnf_g, bnf_b, bufT, nv8, C);
  long nv6 = (long)BATCH * 6 * 6 * 6;
  k_pool_mask<<<dim3((unsigned)((nv6 + 255) / 256)), dim3(256), 0, stream>>>(
      mask, maskO, 8, 6, 3, 1);
  conv(bufT, w_last, maskO, bufH, 8, 6, 128, 256, 3);
  bn(bufH, maskO, bnl_g, bnl_b, bufT, nv6, 256);
  k_mean<<<dim3(2), dim3(256), 0, stream>>>(bufT, pooled, 216, 256);
  k_linear<<<dim3(1), dim3(32), 0, stream>>>(pooled, lin_w, lin_b, (float*)d_out, 256);
}
